// HeterogeneousGatingNetwork_82145544503972
// MI455X (gfx1250) — matrix operand order repaired and checked by reading, not compiled, not run on hardware
//
#include <hip/hip_runtime.h>
#include <hip/hip_bf16.h>
#include <math.h>

// ---------------------------------------------------------------------------
// HeterogeneousGatingNetwork for MI455X (gfx1250, wave32, WMMA bf16)
//   V=50257, D=1024, E=8, B=2048, S=256
// Phase 1 (bandwidth-bound): embedding gather+pool, ~2.15 GB @ 23.3 TB/s.
// Phase 2 (tiny math): ~24 GFLOP of GEMMs via v_wmma_f32_16x16x32_bf16 with
// double-buffered LDS and async global->LDS copies (ASYNCcnt) for A tiles.
// ---------------------------------------------------------------------------

typedef __attribute__((ext_vector_type(16))) __bf16 v16bf;
typedef __attribute__((ext_vector_type(8)))  float  v8f;

#define B_ROWS 2048
#define SEQ    256
#define DMODEL 1024
#define NEXP   8

#define A_STRIDE 40   // halfwords; 80-byte rows -> 16B-aligned async dsts
#define B_STRIDE 34   // halfwords; 17-bank stride, conflict-free

__device__ __forceinline__ unsigned short f2bf(float f) {
    union { float f; unsigned u; } v; v.f = f;
    unsigned r = v.u + 0x7fffu + ((v.u >> 16) & 1u);   // round-to-nearest-even
    return (unsigned short)(r >> 16);
}
__device__ __forceinline__ float bf2f(unsigned short h) {
    union { unsigned u; float f; } v; v.u = ((unsigned)h) << 16;
    return v.f;
}
__device__ __forceinline__ float gelu_exact(float x) {
    return 0.5f * x * (1.0f + erff(x * 0.70710678118654752440f));
}
// Low 32 bits of a flat shared-aperture address are the LDS byte offset
// (CDNA5 ISA 10.2).
__device__ __forceinline__ unsigned lds_off32(const void* p) {
    return (unsigned)(size_t)p;
}

// ---------------------------------------------------------------------------
// Kernel 0: masked mean pooling.  One block per batch row, 256 threads,
// each thread owns 4 consecutive d-columns (float4 coalesced gathers).
// ---------------------------------------------------------------------------
__global__ void __launch_bounds__(256)
pool_kernel(const int* __restrict__ x, const float* __restrict__ emb,
            unsigned short* __restrict__ pooledh)
{
    __shared__ int sidx[SEQ];
    const int b = blockIdx.x;
    const int t = threadIdx.x;
    sidx[t] = x[(size_t)b * SEQ + t];
    __syncthreads();

    float4 acc = make_float4(0.f, 0.f, 0.f, 0.f);
    float  cnt = 0.f;
    const int d0 = t * 4;
    for (int s = 0; s < SEQ; ++s) {
        const int idx = sidx[s];                 // scalar-uniform across the wave
        if (idx != 0) {
            const float4 e = *(const float4*)(emb + (size_t)idx * DMODEL + d0);
            acc.x += e.x; acc.y += e.y; acc.z += e.z; acc.w += e.w;
            cnt += 1.f;
        }
    }
    const float inv = 1.f / (cnt + 1e-8f);
    const unsigned p0 = ((unsigned)f2bf(acc.y * inv) << 16) | f2bf(acc.x * inv);
    const unsigned p1 = ((unsigned)f2bf(acc.w * inv) << 16) | f2bf(acc.z * inv);
    *(uint2*)(pooledh + (size_t)b * DMODEL + d0) = make_uint2(p0, p1);
}

// ---------------------------------------------------------------------------
// Double-buffered WMMA tile: block tile 64(M) x 32(N), BK=32, 256 threads
// (8 waves), one 16x16 v_wmma_f32_16x16x32_bf16 tile per wave.
//   A tile: bf16, copied global->LDS by GLOBAL_LOAD_ASYNC_TO_LDS_B128
//           (no VGPR staging; tracked by ASYNCcnt).
//   B tile: f32 coalesced load -> bf16 convert -> stored transposed Bs[n][k]
//           so B fragments are contiguous b32 LDS reads.
// Fragment lane maps follow ISA 7.12.2 (16-bit A 16x32; B K-half layout).
// ---------------------------------------------------------------------------
template<int K, int N, bool OUTBF>
__device__ __forceinline__ void gemm_gelu_tile(
    const unsigned short* __restrict__ A,    // [M x K] bf16 row-major
    const float*          __restrict__ Bw,   // [K x N] f32 row-major
    const float*          __restrict__ bias, // [N]
    void*                 __restrict__ Out,  // [M x N] bf16 or f32
    unsigned short As[2][64][A_STRIDE], unsigned short Bs[2][32][B_STRIDE])
{
    const int t    = threadIdx.x;
    const int lane = t & 31;
    const int wave = t >> 5;
    const int tm   = wave & 3;          // 4 M-tiles of 16
    const int tn   = wave >> 2;         // 2 N-tiles of 16
    const int bm   = blockIdx.y * 64;
    const int bn   = blockIdx.x * 32;

    const int am  = t >> 2;             // 0..63 (A tile row)
    const int ak  = (t & 3) * 8;        // 0,8,16,24
    const int bk  = t >> 3;             // 0..31 (B tile k-row)
    const int bn4 = (t & 7) * 4;        // 0..28

    // Async copy of this thread's 16 bytes of the A tile straight into LDS.
    auto fetchA = [&](int kt, int buf) {
        const unsigned        dst = lds_off32(&As[buf][am][ak]);
        const unsigned short* src = A + (size_t)(bm + am) * K + kt * 32 + ak;
        asm volatile("global_load_async_to_lds_b128 %0, %1, off"
                     :: "v"(dst), "v"(src) : "memory");
    };
    // B tile: 32(K) x 32(N) f32 coalesced, cvt->bf16, store transposed.
    auto fetchB = [&](int kt, int buf) {
        const float4 bv = *(const float4*)(Bw + (size_t)(kt * 32 + bk) * N + bn + bn4);
        Bs[buf][bn4 + 0][bk] = f2bf(bv.x);
        Bs[buf][bn4 + 1][bk] = f2bf(bv.y);
        Bs[buf][bn4 + 2][bk] = f2bf(bv.z);
        Bs[buf][bn4 + 3][bk] = f2bf(bv.w);
    };

    v8f acc = {};
    const int NT = K / 32;

    fetchA(0, 0);
    fetchB(0, 0);

    for (int kt = 0; kt < NT; ++kt) {
        const int cur = kt & 1;
        // Drain this thread's async LDS writes, then make all tile writes
        // visible block-wide (compiler adds s_wait_dscnt for the B stores).
        asm volatile("s_wait_asynccnt 0x0" ::: "memory");
        __syncthreads();

        if (kt + 1 < NT) {              // prefetch next tile into other buffer
            fetchA(kt + 1, cur ^ 1);
            fetchB(kt + 1, cur ^ 1);
        }

        union { unsigned u[8]; v16bf v; } afr, bfr;
        const int arow = tm * 16 + (lane & 15);
        const int half = lane >> 4;
#pragma unroll
        for (int j = 0; j < 8; ++j) {   // A: K = {0..7|8..15} then {16..23|24..31}
            const int ka = ((j & 3) * 2) + half * 8 + ((j >> 2) * 16);
            afr.u[j] = *(const unsigned*)&As[cur][arow][ka];
        }
        const int brow = tn * 16 + (lane & 15);
#pragma unroll
    for (int j = 0; j < 8; ++j) {   // B: K = {0..7|8..15} then {16..23|24..31}
        const int kb = ((j & 3) * 2) + half * 8 + ((j >> 2) * 16);
        bfr.u[j] = *(const unsigned*)&Bs[cur][brow][kb];
    }

        acc = __builtin_amdgcn_wmma_f32_16x16x32_bf16(
                  false, afr.v, false, bfr.v, (short)0, acc, false, false);
    }

    // Epilogue: bias + exact GELU, C/D layout: VGPR i -> M = i + 8*(lane>=16)
    const int   gn = bn + tn * 16 + (lane & 15);
    const float bv = bias[gn];
#pragma unroll
    for (int i = 0; i < 8; ++i) {
        const int   gm = bm + tm * 16 + i + 8 * (lane >> 4);
        const float v  = gelu_exact(acc[i] + bv);
        if (OUTBF)
            ((unsigned short*)Out)[(size_t)gm * N + gn] = f2bf(v);
        else
            ((float*)Out)[(size_t)gm * N + gn] = v;
    }
}

// GEMM 1: z in [0,8): h1[z] = gelu(pooled @ w1[z] + b1[z])
//         z == 8    : g1    = gelu(pooled @ gw1   + gb1)      (same shape!)
__global__ void __launch_bounds__(256)
gemm1_kernel(const unsigned short* __restrict__ pooledh,
             const float* __restrict__ w1, const float* __restrict__ b1,
             const float* __restrict__ gw1, const float* __restrict__ gb1,
             unsigned short* __restrict__ h1, unsigned short* __restrict__ g1)
{
    __shared__ unsigned short As[2][64][A_STRIDE];
    __shared__ unsigned short Bs[2][32][B_STRIDE];
    const int z = blockIdx.z;
    const float*    Bw  = (z < NEXP) ? (w1 + (size_t)z * DMODEL * 512) : gw1;
    const float*    bia = (z < NEXP) ? (b1 + z * 512) : gb1;
    unsigned short* out = (z < NEXP) ? (h1 + (size_t)z * B_ROWS * 512) : g1;
    gemm_gelu_tile<DMODEL, 512, true>(pooledh, Bw, bia, out, As, Bs);
}

// GEMM 2: h2[e] = gelu(h1[e] @ w2[e] + b2[e]),  2048x512x256, fp32 out
__global__ void __launch_bounds__(256)
gemm2_kernel(const unsigned short* __restrict__ h1,
             const float* __restrict__ w2, const float* __restrict__ b2,
             float* __restrict__ h2)
{
    __shared__ unsigned short As[2][64][A_STRIDE];
    __shared__ unsigned short Bs[2][32][B_STRIDE];
    const int e = blockIdx.z;
    gemm_gelu_tile<512, 256, false>(h1 + (size_t)e * B_ROWS * 512,
                                    w2 + (size_t)e * 512 * 256,
                                    b2 + e * 256,
                                    h2 + (size_t)e * B_ROWS * 256, As, Bs);
}

// ---------------------------------------------------------------------------
// Kernel 3: per-row expert scores + global-gate scores + softmax.
// Wave e (of 8) computes expert e's dot(h2[e,b,:], wg[e]) and the global
// gate column dot(g1[b,:], gw2[:,e]); wave32 shuffle reduction.
// ---------------------------------------------------------------------------
__global__ void __launch_bounds__(256)
score_softmax_kernel(const float* __restrict__ h2, const unsigned short* __restrict__ g1,
                     const float* __restrict__ wg, const float* __restrict__ bg,
                     const float* __restrict__ gw2, const float* __restrict__ gb2,
                     float* __restrict__ out)
{
    __shared__ float comb[NEXP];
    const int b    = blockIdx.x;
    const int lane = threadIdx.x & 31;
    const int e    = threadIdx.x >> 5;   // 0..7

    float s = 0.f;
    const float* h2b = h2 + ((size_t)e * B_ROWS + b) * 256;
    const float* wge = wg + e * 256;
    for (int q = lane; q < 256; q += 32) s += h2b[q] * wge[q];

    const unsigned short* g1b = g1 + (size_t)b * 512;
    for (int q = lane; q < 512; q += 32) s += bf2f(g1b[q]) * gw2[q * NEXP + e];

#pragma unroll
    for (int off = 16; off > 0; off >>= 1) s += __shfl_down(s, off, 32);
    if (lane == 0) comb[e] = s + bg[e] + gb2[e];
    __syncthreads();

    if (threadIdx.x == 0) {
        float m = comb[0];
        for (int i = 1; i < NEXP; ++i) m = fmaxf(m, comb[i]);
        float ex[NEXP], sum = 0.f;
        for (int i = 0; i < NEXP; ++i) { ex[i] = expf(comb[i] - m); sum += ex[i]; }
        const float inv = 1.f / sum;
        for (int i = 0; i < NEXP; ++i) out[(size_t)b * NEXP + i] = ex[i] * inv;
    }
}

// ---------------------------------------------------------------------------
// Kernel 4: load-balance loss = unbiased var(mean_b gate_weights) * E
// ---------------------------------------------------------------------------
__global__ void __launch_bounds__(256)
loss_kernel(const float* __restrict__ gwts, float* __restrict__ out)
{
    __shared__ float red[256][NEXP];
    const int t = threadIdx.x;
    float loc[NEXP];
#pragma unroll
    for (int e = 0; e < NEXP; ++e) loc[e] = 0.f;
    for (int r = t; r < B_ROWS; r += 256)
#pragma unroll
        for (int e = 0; e < NEXP; ++e) loc[e] += gwts[(size_t)r * NEXP + e];
#pragma unroll
    for (int e = 0; e < NEXP; ++e) red[t][e] = loc[e];
    __syncthreads();
    for (int st = 128; st > 0; st >>= 1) {
        if (t < st)
#pragma unroll
            for (int e = 0; e < NEXP; ++e) red[t][e] += red[t + st][e];
        __syncthreads();
    }
    if (t == 0) {
        float u[NEXP], mean = 0.f;
        for (int e = 0; e < NEXP; ++e) { u[e] = red[0][e] / (float)B_ROWS; mean += u[e]; }
        mean /= (float)NEXP;
        float var = 0.f;
        for (int e = 0; e < NEXP; ++e) { const float d = u[e] - mean; var += d * d; }
        var /= (float)(NEXP - 1);                 // ddof=1
        out[B_ROWS * NEXP] = var * (float)NEXP;
    }
}

// ---------------------------------------------------------------------------
extern "C" void kernel_launch(void* const* d_in, const int* in_sizes, int n_in,
                              void* d_out, int out_size, void* d_ws, size_t ws_size,
                              hipStream_t stream)
{
    const int*   x   = (const int*)  d_in[0];
    const float* emb = (const float*)d_in[1];
    const float* w1  = (const float*)d_in[2];
    const float* b1  = (const float*)d_in[3];
    const float* w2  = (const float*)d_in[4];
    const float* b2  = (const float*)d_in[5];
    const float* wg  = (const float*)d_in[6];
    const float* bg  = (const float*)d_in[7];
    const float* gw1 = (const float*)d_in[8];
    const float* gb1 = (const float*)d_in[9];
    const float* gw2 = (const float*)d_in[10];
    const float* gb2 = (const float*)d_in[11];
    float* out = (float*)d_out;

    char* ws = (char*)d_ws;
    unsigned short* pooledh = (unsigned short*)(ws);                 // 4 MB  [2048,1024] bf16
    unsigned short* h1      = (unsigned short*)(ws + (4u  << 20));   // 16 MB [8,2048,512] bf16
    unsigned short* g1      = (unsigned short*)(ws + (20u << 20));   // 2 MB  [2048,512]   bf16
    float*          h2      = (float*)         (ws + (22u << 20));   // 16 MB [8,2048,256] f32

    pool_kernel<<<dim3(B_ROWS), dim3(256), 0, stream>>>(x, emb, pooledh);
    gemm1_kernel<<<dim3(512 / 32, B_ROWS / 64, NEXP + 1), dim3(256), 0, stream>>>(
        pooledh, w1, b1, gw1, gb1, h1, g1);
    gemm2_kernel<<<dim3(256 / 32, B_ROWS / 64, NEXP), dim3(256), 0, stream>>>(
        h1, w2, b2, h2);
    score_softmax_kernel<<<dim3(B_ROWS), dim3(256), 0, stream>>>(
        h2, g1, wg, bg, gw2, gb2, out);
    loss_kernel<<<dim3(1), dim3(256), 0, stream>>>(out, out);
}
